// AttentionNetLayer_35270271434975
// MI455X (gfx1250) — compile-verified, matrix operand order checked by reading, not run on hardware
//
#include <hip/hip_runtime.h>
#include <math.h>

#define BATCH   2
#define NPTS    8192
#define NPOINT  2048
#define NSAMPLE 32
#define C_IN    64
#define D_IN    67      // 3 + C_IN
#define D_PAD   96      // padded K-dim: 3 chunks of 32
#define KD      128     // KEY_DIM
#define OD      128     // OUT_DIM
#define RAD2    0.04f   // RADIUS^2
#define WAVES   8       // waves per block in wave-per-centroid kernels
#define NFRAG   24      // 8 N-tiles x 3 K-chunks
#define FRAGSZ  512     // 32 lanes x 16 halves per fragment

typedef _Float16 half_t;
typedef half_t v16h __attribute__((ext_vector_type(16)));
typedef float  v8f  __attribute__((ext_vector_type(8)));

// ---------------------------------------------------------------------------
// Kernel 1: Farthest Point Sampling. One block per batch; xyz + running min
// distance array live in LDS (128KB of the 320KB WGP LDS).
// ---------------------------------------------------------------------------
__global__ __launch_bounds__(1024) void fps_kernel(
    const float* __restrict__ xyz, float* __restrict__ new_xyz,
    int* __restrict__ fps_idx) {
  const int b = blockIdx.x;
  extern __shared__ char fps_sm[];
  float* sx   = (float*)fps_sm;
  float* sy   = sx + NPTS;
  float* sz   = sy + NPTS;
  float* dist = sz + NPTS;
  __shared__ float rval[32];
  __shared__ int   rind[32];
  __shared__ int   s_far;

  const float* xb = xyz + (size_t)b * NPTS * 3;
  for (int i = threadIdx.x; i < NPTS; i += blockDim.x) {
    sx[i] = xb[i * 3 + 0];
    sy[i] = xb[i * 3 + 1];
    sz[i] = xb[i * 3 + 2];
    dist[i] = 1e10f;
  }
  if (threadIdx.x == 0) s_far = 0;
  __syncthreads();

  for (int it = 0; it < NPOINT; ++it) {
    const int far = s_far;
    const float cx = sx[far], cy = sy[far], cz = sz[far];
    if (threadIdx.x == 0) {
      fps_idx[b * NPOINT + it] = far;
      new_xyz[((size_t)b * NPOINT + it) * 3 + 0] = cx;
      new_xyz[((size_t)b * NPOINT + it) * 3 + 1] = cy;
      new_xyz[((size_t)b * NPOINT + it) * 3 + 2] = cz;
    }
    float best = -1.0f;
    int   besti = 0;
    for (int i = threadIdx.x; i < NPTS; i += blockDim.x) {
      const float dx = sx[i] - cx, dy = sy[i] - cy, dz = sz[i] - cz;
      float d = fminf(dist[i], dx * dx + dy * dy + dz * dz);
      dist[i] = d;
      if (d > best) { best = d; besti = i; }
    }
    // wave32 argmax (tie -> lowest index, matching jnp.argmax)
    #pragma unroll
    for (int off = 16; off > 0; off >>= 1) {
      const float ov = __shfl_xor(best, off);
      const int   oi = __shfl_xor(besti, off);
      if (ov > best || (ov == best && oi < besti)) { best = ov; besti = oi; }
    }
    const int wid = threadIdx.x >> 5;
    if ((threadIdx.x & 31) == 0) { rval[wid] = best; rind[wid] = besti; }
    __syncthreads();
    if (threadIdx.x < 32) {
      const int nw = blockDim.x >> 5;
      float v  = (threadIdx.x < nw) ? rval[threadIdx.x] : -1.0f;
      int   ix = (threadIdx.x < nw) ? rind[threadIdx.x] : 0x7fffffff;
      #pragma unroll
      for (int off = 16; off > 0; off >>= 1) {
        const float ov = __shfl_xor(v, off);
        const int   oi = __shfl_xor(ix, off);
        if (ov > v || (ov == v && oi < ix)) { v = ov; ix = oi; }
      }
      if (threadIdx.x == 0) s_far = ix;
    }
    __syncthreads();
  }
}

// ---------------------------------------------------------------------------
// Kernel 2: Ball query. One wave32 per centroid; ballot + prefix-popcount
// collects the first NSAMPLE in-radius indices in index order.
// ---------------------------------------------------------------------------
__global__ __launch_bounds__(256) void ballq_kernel(
    const float* __restrict__ xyz, const float* __restrict__ new_xyz,
    int* __restrict__ idx_out) {
  const int wave = threadIdx.x >> 5;
  const int lane = threadIdx.x & 31;
  const int gw = blockIdx.x * WAVES + wave;            // b*NPOINT + m
  const int b = gw / NPOINT;
  __shared__ int slots[WAVES][NSAMPLE];
  int* slot = slots[wave];

  const float* xb = xyz + (size_t)b * NPTS * 3;
  const float cx = new_xyz[(size_t)gw * 3 + 0];
  const float cy = new_xyz[(size_t)gw * 3 + 1];
  const float cz = new_xyz[(size_t)gw * 3 + 2];

  int count = 0;
  for (int base = 0; base < NPTS && count < NSAMPLE; base += 32) {
    const int i = base + lane;
    const float dx = xb[i * 3 + 0] - cx;
    const float dy = xb[i * 3 + 1] - cy;
    const float dz = xb[i * 3 + 2] - cz;
    const bool within = (dx * dx + dy * dy + dz * dz) < RAD2;
    const unsigned long long mask = __ballot(within);
    const int pos = count + __popcll(mask & ((1ull << lane) - 1ull));
    if (within && pos < NSAMPLE) slot[pos] = i;
    count += __popcll(mask);
  }
  asm volatile("s_wait_dscnt 0" ::: "memory");
  int val;
  if (count == 0) {
    val = NPTS - 1;                                    // clip(N, 0, N-1)
  } else {
    const int first = slot[0];
    val = (lane < count) ? slot[lane] : first;
  }
  idx_out[(size_t)gw * NSAMPLE + lane] = val;
}

// ---------------------------------------------------------------------------
// Kernel 3: fused group + K/V projection (WMMA f16->f32) + attention.
// One wave32 per centroid; 96 v_wmma_f32_16x16x32_f16 per centroid.
// Weights are pre-swizzled into B-fragment layout so each lane's fragment is
// one contiguous 32B v16h load (2x ds_load_b128).
// ---------------------------------------------------------------------------
__device__ __forceinline__ int kmap(int lane, int i) {
  // ISA 16-bit A/B packing: lanes 0-15 hold K {0..7,16..23}, lanes 16-31 hold
  // K {8..15,24..31} within a 32-wide K chunk.
  return ((lane < 16) ? 0 : 8) + ((i < 8) ? i : i + 8);
}

__global__ __launch_bounds__(256) void attn_kernel(
    const float* __restrict__ xyz, const float* __restrict__ points,
    const float* __restrict__ Wq, const float* __restrict__ bq,
    const float* __restrict__ Wk, const float* __restrict__ bk,
    const float* __restrict__ Wv, const float* __restrict__ bv,
    const float* __restrict__ new_xyz, const int* __restrict__ idx,
    float* __restrict__ out) {
  extern __shared__ char dynsm[];
  half_t* sWkF = (half_t*)dynsm;                 // [NFRAG][32][16] fragments
  half_t* sWvF = sWkF + NFRAG * FRAGSZ;          // [NFRAG][32][16]
  half_t* sFt  = sWvF + NFRAG * FRAGSZ;          // [WAVES][32][D_PAD]
  float*  sQ   = (float*)(sFt + WAVES * 32 * D_PAD);   // [WAVES][KD]
  float*  sSc  = sQ + WAVES * KD;                // [WAVES][NSAMPLE]

  // Stage weights pre-swizzled into WMMA B-fragment layout (zero-padded K).
  for (int t = threadIdx.x; t < NFRAG * FRAGSZ; t += blockDim.x) {
    const int frag = t >> 9;            // Nt*3 + c
    const int rem  = t & 511;
    const int ln   = rem >> 4;
    const int i    = rem & 15;
    const int Nt = frag / 3, c = frag % 3;
    const int k = c * 32 + kmap(ln, i);
    const int n = Nt * 16 + (ln & 15);
    sWkF[t] = (half_t)((k < D_IN) ? Wk[k * KD + n] : 0.0f);
    sWvF[t] = (half_t)((k < D_IN) ? Wv[k * OD + n] : 0.0f);
  }
  __syncthreads();

  const int wave = threadIdx.x >> 5;
  const int lane = threadIdx.x & 31;
  const int gw = blockIdx.x * WAVES + wave;     // b*NPOINT + m
  const int b = gw / NPOINT;
  const float cx = new_xyz[(size_t)gw * 3 + 0];
  const float cy = new_xyz[(size_t)gw * 3 + 1];
  const float cz = new_xyz[(size_t)gw * 3 + 2];

  // Stage grouped feature tile: lane == sample s, padded row of 96 halves.
  {
    const int p = idx[(size_t)gw * NSAMPLE + lane];
    const float* xr = xyz + ((size_t)b * NPTS + p) * 3;
    const float* pr = points + ((size_t)b * NPTS + p) * C_IN;
    half_t* fr = sFt + (wave * 32 + lane) * D_PAD;
    fr[0] = (half_t)(xr[0] - cx);
    fr[1] = (half_t)(xr[1] - cy);
    fr[2] = (half_t)(xr[2] - cz);
    #pragma unroll
    for (int d = 0; d < C_IN; ++d) fr[3 + d] = (half_t)pr[d];
    #pragma unroll
    for (int d = D_IN; d < D_PAD; ++d) fr[d] = (half_t)0.0f;
  }

  // Q in f32 (3x128 matvec), pre-scaled by 1/sqrt(KD); zero scores.
  {
    const float scale = 0.088388347648318447f;  // 1/sqrt(128)
    #pragma unroll
    for (int r = 0; r < 4; ++r) {
      const int d = lane + 32 * r;
      const float q = bq[d] + cx * Wq[0 * KD + d] + cy * Wq[1 * KD + d] +
                      cz * Wq[2 * KD + d];
      sQ[wave * KD + d] = q * scale;
    }
    sSc[wave * NSAMPLE + lane] = 0.0f;
  }
  asm volatile("s_wait_dscnt 0" ::: "memory");

  // A fragments (2 M-tiles x 3 K-chunks) held in VGPRs, reused for K and V.
  v16h afrag[2][3];
  #pragma unroll
  for (int Mt = 0; Mt < 2; ++Mt) {
    #pragma unroll
    for (int c = 0; c < 3; ++c) {
      const half_t* row =
          sFt + (wave * 32 + Mt * 16 + (lane & 15)) * D_PAD + c * 32;
      v16h a;
      #pragma unroll
      for (int i = 0; i < 16; ++i) a[i] = row[kmap(lane, i)];
      afrag[Mt][c] = a;
    }
  }

  const int sbaseHi = (lane < 16) ? 0 : 8;      // C/D row offset for lane group

  // Pass 1: K = tile @ Wk + bk via WMMA. The (Mt,j)->sample mapping is
  // Nt-invariant, so partial scores accumulate in registers across Nt and are
  // flushed with 16 ds_add_f32 at the end.
  float scAcc[2][8];
  #pragma unroll
  for (int Mt = 0; Mt < 2; ++Mt)
    #pragma unroll
    for (int j = 0; j < 8; ++j) scAcc[Mt][j] = 0.0f;

  for (int Nt = 0; Nt < 8; ++Nt) {
    const float bkv = bk[Nt * 16 + (lane & 15)];
    const float q = sQ[wave * KD + Nt * 16 + (lane & 15)];
    #pragma unroll
    for (int Mt = 0; Mt < 2; ++Mt) {
      v8f acc;
      #pragma unroll
      for (int j = 0; j < 8; ++j) acc[j] = bkv;
      #pragma unroll
      for (int c = 0; c < 3; ++c) {
        const v16h bf =
            *(const v16h*)(sWkF + (Nt * 3 + c) * FRAGSZ + lane * 16);
        acc = __builtin_amdgcn_wmma_f32_16x16x32_f16(
            false, afrag[Mt][c], false, bf, (short)0, acc, false, false);
      }
      #pragma unroll
      for (int j = 0; j < 8; ++j) scAcc[Mt][j] += q * acc[j];
    }
  }
  #pragma unroll
  for (int Mt = 0; Mt < 2; ++Mt)
    #pragma unroll
    for (int j = 0; j < 8; ++j)
      atomicAdd(&sSc[wave * NSAMPLE + Mt * 16 + sbaseHi + j], scAcc[Mt][j]);
  asm volatile("s_wait_dscnt 0" ::: "memory");

  // Softmax over the 32 samples (lane == sample).
  {
    const float sc = sSc[wave * NSAMPLE + lane];
    float mx = sc;
    #pragma unroll
    for (int off = 16; off > 0; off >>= 1) mx = fmaxf(mx, __shfl_xor(mx, off));
    const float e = __expf(sc - mx);
    float sum = e;
    #pragma unroll
    for (int off = 16; off > 0; off >>= 1) sum += __shfl_xor(sum, off);
    sSc[wave * NSAMPLE + lane] = e / sum;
  }
  asm volatile("s_wait_dscnt 0" ::: "memory");

  // Pass 2: V = tile @ Wv + bv via WMMA, weighted sum over samples with the
  // 16 softmax weights hoisted into registers.
  float wreg[2][8];
  #pragma unroll
  for (int Mt = 0; Mt < 2; ++Mt)
    #pragma unroll
    for (int j = 0; j < 8; ++j)
      wreg[Mt][j] = sSc[wave * NSAMPLE + Mt * 16 + sbaseHi + j];

  for (int Nt = 0; Nt < 8; ++Nt) {
    const float bvv = bv[Nt * 16 + (lane & 15)];
    float colsum = 0.0f;
    #pragma unroll
    for (int Mt = 0; Mt < 2; ++Mt) {
      v8f acc;
      #pragma unroll
      for (int j = 0; j < 8; ++j) acc[j] = bvv;
      #pragma unroll
      for (int c = 0; c < 3; ++c) {
        const v16h bf =
            *(const v16h*)(sWvF + (Nt * 3 + c) * FRAGSZ + lane * 16);
        acc = __builtin_amdgcn_wmma_f32_16x16x32_f16(
            false, afrag[Mt][c], false, bf, (short)0, acc, false, false);
      }
      #pragma unroll
      for (int j = 0; j < 8; ++j) colsum += wreg[Mt][j] * acc[j];
    }
    colsum += __shfl_xor(colsum, 16);           // fold the two lane halves
    if (lane < 16) out[(size_t)gw * OD + Nt * 16 + lane] = colsum;
  }
}

// ---------------------------------------------------------------------------
extern "C" void kernel_launch(void* const* d_in, const int* in_sizes, int n_in,
                              void* d_out, int out_size, void* d_ws,
                              size_t ws_size, hipStream_t stream) {
  (void)in_sizes; (void)n_in; (void)out_size; (void)ws_size;
  const float* xyz    = (const float*)d_in[0];
  const float* points = (const float*)d_in[1];
  const float* Wq     = (const float*)d_in[2];
  const float* bq     = (const float*)d_in[3];
  const float* Wk     = (const float*)d_in[4];
  const float* bk     = (const float*)d_in[5];
  const float* Wv     = (const float*)d_in[6];
  const float* bv     = (const float*)d_in[7];

  // Output tuple layout: new_xyz [B,2048,3] f32 | out [B,2048,128] f32 |
  // idx [B,2048,32] i32 (stored as raw int bits in the tail).
  float* out_f    = (float*)d_out;
  float* new_xyz  = out_f;
  float* out_attn = out_f + (size_t)BATCH * NPOINT * 3;
  int*   idx      = (int*)(out_attn + (size_t)BATCH * NPOINT * OD);
  int*   fps_idx  = (int*)d_ws;                 // B*NPOINT ints of scratch

  const size_t fps_lds = (size_t)4 * NPTS * sizeof(float);      // 128 KB
  fps_kernel<<<BATCH, 1024, fps_lds, stream>>>(xyz, new_xyz, fps_idx);

  ballq_kernel<<<(BATCH * NPOINT) / WAVES, 256, 0, stream>>>(xyz, new_xyz, idx);

  const size_t attn_lds =
      (size_t)(2 * NFRAG * FRAGSZ + WAVES * 32 * D_PAD) * sizeof(half_t) +
      (size_t)(WAVES * KD + WAVES * NSAMPLE) * sizeof(float);   // ~101 KB
  attn_kernel<<<(BATCH * NPOINT) / WAVES, 256, attn_lds, stream>>>(
      xyz, points, Wq, bq, Wk, bk, Wv, bv, new_xyz, idx, out_attn);
}
